// MCPHiddenLayers_31284541784771
// MI455X (gfx1250) — compile-verified, matrix operand order checked by reading, not run on hardware
//
#include <hip/hip_runtime.h>

// ---------------------------------------------------------------------------
// CDNA5 (gfx1250) bf16 WMMA implementation of the MCP hidden-layer network.
// wave32; V_WMMA_F32_16X16X32_BF16 (f32 accumulate), fused epilogues,
// async global->LDS tile staging (ASYNCcnt) with double buffering.
// ---------------------------------------------------------------------------

typedef __attribute__((ext_vector_type(16))) __bf16 v16bf;
typedef __attribute__((ext_vector_type(8)))  float  v8f;
typedef __attribute__((address_space(3)))    __bf16 lds_bf16_t;

__device__ __forceinline__ v8f wmma_bf16(v16bf a, v16bf b, v8f c) {
  // 8 args: (neg_a, A, neg_b, B, c_mod, C, reuse_a, reuse_b)
  return __builtin_amdgcn_wmma_f32_16x16x32_bf16(false, a, false, b,
                                                 (short)0, c, false, false);
}

union FragU { uint4 u[2]; v16bf v; };

__device__ __forceinline__ v16bf ld_frag(const __bf16* p) {
  FragU f;
  f.u[0] = *(const uint4*)(p);
  f.u[1] = *(const uint4*)(p + 8);
  return f.v;
}

// Async copy of 16 bytes global -> LDS (GLOBAL_LOAD_ASYNC_TO_LDS_B128,
// tracked with ASYNCcnt; no VGPR round-trip).
__device__ __forceinline__ void async_cp16(__bf16* lds_dst, const __bf16* gsrc) {
  unsigned loff = (unsigned)(size_t)(lds_bf16_t*)lds_dst;
  asm volatile("global_load_async_to_lds_b128 %0, %1, off"
               :: "v"(loff), "v"(gsrc) : "memory");
}

__device__ __forceinline__ void wait_async0() {
  asm volatile("s_wait_asynccnt 0" ::: "memory");
}

// ---------------------------------------------------------------------------
// f32 -> bf16 straight conversion (features)
// ---------------------------------------------------------------------------
__global__ __launch_bounds__(256)
void f32_to_bf16(const float* __restrict__ src, __bf16* __restrict__ dst, size_t n) {
  size_t i = ((size_t)blockIdx.x * 256 + threadIdx.x) * 4;
  if (i + 3 < n) {
    float4 v = *(const float4*)(src + i);
    dst[i + 0] = (__bf16)v.x;
    dst[i + 1] = (__bf16)v.y;
    dst[i + 2] = (__bf16)v.z;
    dst[i + 3] = (__bf16)v.w;
  } else {
    for (; i < n; ++i) dst[i] = (__bf16)src[i];
  }
}

// ---------------------------------------------------------------------------
// f32 (K x N row-major) -> bf16 (N x K row-major) transpose-convert.
// blockIdx.z batches matrices (for pw1/pw2).
// ---------------------------------------------------------------------------
__global__ __launch_bounds__(256)
void f32_transpose_to_bf16(const float* __restrict__ src, __bf16* __restrict__ dst,
                           int K, int N) {
  size_t mat = blockIdx.z;
  const float* s = src + mat * (size_t)K * N;
  __bf16*      d = dst + mat * (size_t)K * N;
  int i = blockIdx.x * 256 + threadIdx.x;
  if (i < K * N) {
    int k = i / N, n = i - k * N;
    d[(size_t)n * K + k] = (__bf16)s[i];
  }
}

// ---------------------------------------------------------------------------
// Generic fused GEMM: out[m,n] = act( sum_k A[m,k]*Wt[n,k] + bias[n] )
//   A : M x K bf16, row stride lda
//   Wt: N x K bf16 row-major (pre-transposed weights; L2-resident)
// Tile: BM=128 x BN=64, BK=32; 8 waves, wave w owns rows [w*16,w*16+16) and
// all 4 N-subtiles. Both operands staged in LDS by async global->LDS copies,
// double buffered so tile s+1 streams in while tile s runs on the WMMA units.
// ACT: 1 = relu, 2 = sigmoid.  OB: bf16 output (else f32).
// ---------------------------------------------------------------------------
template <int ACT, bool OB>
__global__ __launch_bounds__(256)
void gemm_bias_act(const __bf16* __restrict__ A, int lda,
                   const __bf16* __restrict__ Wt,
                   const float* __restrict__ bias,
                   void* __restrict__ outp, int ldo,
                   int M, int N, int K) {
  constexpr int BM = 128, BN = 64, BK = 32, PAD = 8, PITCH = BK + PAD;
  __shared__ __bf16 sA[2][BM][PITCH];
  __shared__ __bf16 sB[2][BN][PITCH];

  const int tid   = threadIdx.x;
  const int wave  = tid >> 5;
  const int lane  = tid & 31;
  const int lrow  = lane & 15;
  const int khalf = lane >> 4;
  const int m0    = blockIdx.y * BM;
  const int n0    = blockIdx.x * BN;
  const int mw    = wave * 16;

  // staging coords: A tile 128x32 (all 256 threads, 16 bf16 each),
  //                 B tile 64x32 (threads 128..255, 16 bf16 each)
  const int ar = tid >> 1;
  const int ac = (tid & 1) * 16;
  const int br = (tid - 128) >> 1;
  const int bc = ((tid - 128) & 1) * 16;

  auto stage = [&](int k0, int buf) {
    const __bf16* gA = A + (size_t)(m0 + ar) * lda + k0 + ac;
    async_cp16(&sA[buf][ar][ac],     gA);
    async_cp16(&sA[buf][ar][ac + 8], gA + 8);
    if (tid >= 128) {
      if (n0 + br < N) {
        const __bf16* gB = Wt + (size_t)(n0 + br) * K + k0 + bc;
        async_cp16(&sB[buf][br][bc],     gB);
        async_cp16(&sB[buf][br][bc + 8], gB + 8);
      }
    }
  };

  v8f acc[4] = {};

  stage(0, 0);
  wait_async0();
  __syncthreads();

  const int ksteps = K / BK;
  for (int s = 0; s < ksteps; ++s) {
    const int cur = s & 1;
    if (s + 1 < ksteps) stage((s + 1) * BK, cur ^ 1);  // overlap with WMMA

    v16bf af = ld_frag(&sA[cur][mw + lrow][khalf * 16]);
#pragma unroll
    for (int t = 0; t < 4; ++t) {
      if (n0 + t * 16 < N) {
        v16bf bf = ld_frag(&sB[cur][t * 16 + lrow][khalf * 16]);
        acc[t] = wmma_bf16(af, bf, acc[t]);
      }
    }
    wait_async0();
    __syncthreads();
  }

  // Epilogue. C layout: VGPR r, lane l -> (M = r + 8*(l>>4), N = l&15)
#pragma unroll
  for (int t = 0; t < 4; ++t) {
    int nt = n0 + t * 16;
    if (nt < N) {
      int n = nt + lrow;
      float bb = bias[n];
#pragma unroll
      for (int r = 0; r < 8; ++r) {
        size_t m = (size_t)m0 + mw + khalf * 8 + r;
        float v = acc[t][r] + bb;
        if (ACT == 1) v = fmaxf(v, 0.f);
        if (ACT == 2) v = 1.f / (1.f + __expf(-v));
        if (OB) ((__bf16*)outp)[m * ldo + n] = (__bf16)v;
        else    ((float*)outp)[m * ldo + n]  = v;
      }
    }
  }
}

// ---------------------------------------------------------------------------
// Fused primitive mixture head. Per 32-row batch tile:
//   for p in 0..15:
//     h   = relu(pe(32x256) @ pw1t[p] + pb1[p])        (kept in LDS, bf16)
//     o   = h @ pw2t[p] + pb2[p]                       (32x64, LDS f32)
//     inv = w[b,p] * exp(-o[:,32:64]); accumulate inv, inv*mu
//   mean = sum(inv*mu)/sum(inv)
// Avoids materializing the (B,16,256) h tensor (~536 MB of traffic).
// Weight fragments come straight from global (L2-resident, interleaves with
// the WMMAs per round-1 codegen).
// ---------------------------------------------------------------------------
__global__ __launch_bounds__(256)
void prim_head(const __bf16* __restrict__ pe,    // B x 256
               const __bf16* __restrict__ pw1t,  // P x (256 x 256), N-major
               const float*  __restrict__ pb1,   // P x 256
               const __bf16* __restrict__ pw2t,  // P x (64 x 256), N-major
               const float*  __restrict__ pb2,   // P x 64
               const float*  __restrict__ wgt,   // B x 16
               float* __restrict__ mean)         // B x 32
{
  constexpr int BM = 32, H = 256, PAD = 8, AOUT = 32;
  __shared__ __bf16 sPE[BM][H + PAD];
  __shared__ __bf16 sH[BM][H + PAD];
  __shared__ float  sO[BM][64 + PAD];
  __shared__ float  sInv[BM][AOUT];
  __shared__ float  sInvMu[BM][AOUT];
  __shared__ float  sW[BM][16];

  const int tid   = threadIdx.x;
  const int wave  = tid >> 5;
  const int lane  = tid & 31;
  const int lrow  = lane & 15;
  const int khalf = lane >> 4;
  const int b0    = blockIdx.x * BM;

  // stage prim_embed tile: 32 x 256 bf16, 32 elems / thread (async)
  {
    int r = tid >> 3, c = (tid & 7) * 32;
    const __bf16* g = pe + (size_t)(b0 + r) * H + c;
#pragma unroll
    for (int q = 0; q < 4; ++q)
      async_cp16(&sPE[r][c + q * 8], g + q * 8);
  }
  // stage gate weights 32 x 16 f32
  {
    int i = tid;
    sW[i >> 4][i & 15] = wgt[(size_t)(b0 + (i >> 4)) * 16 + (i & 15)];
    int j = tid + 256;
    sW[j >> 4][j & 15] = wgt[(size_t)(b0 + (j >> 4)) * 16 + (j & 15)];
  }
  for (int i = tid; i < BM * AOUT; i += 256) {
    sInv[i >> 5][i & 31]   = 0.f;
    sInvMu[i >> 5][i & 31] = 0.f;
  }
  wait_async0();
  __syncthreads();

  for (int p = 0; p < 16; ++p) {
    // ---- GEMM1: 32x256 output; waves 2(M) x 4(N), 4 WMMA tiles each ----
    {
      const int mw = (wave & 1) * 16;
      const int nb = (wave >> 1) * 64;
      const __bf16* W = pw1t + (size_t)p * 256 * 256;
      v8f acc[4] = {};
      for (int k0 = 0; k0 < 256; k0 += 32) {
        v16bf af = ld_frag(&sPE[mw + lrow][k0 + khalf * 16]);
#pragma unroll
        for (int t = 0; t < 4; ++t) {
          v16bf bf = ld_frag(W + (size_t)(nb + t * 16 + lrow) * 256 + k0 + khalf * 16);
          acc[t] = wmma_bf16(af, bf, acc[t]);
        }
      }
#pragma unroll
      for (int t = 0; t < 4; ++t) {
        int n = nb + t * 16 + lrow;
        float bb = pb1[p * 256 + n];
#pragma unroll
        for (int r = 0; r < 8; ++r) {
          int m = mw + khalf * 8 + r;
          sH[m][n] = (__bf16)fmaxf(acc[t][r] + bb, 0.f);
        }
      }
    }
    __syncthreads();

    // ---- GEMM2: 32x64 output; 8 waves x 1 WMMA tile ----
    {
      const int mw = (wave >> 2) * 16;
      const int nb = (wave & 3) * 16;
      const __bf16* W = pw2t + (size_t)p * 64 * 256;
      v8f acc = {};
      for (int k0 = 0; k0 < 256; k0 += 32) {
        v16bf af = ld_frag(&sH[mw + lrow][k0 + khalf * 16]);
        v16bf bf = ld_frag(W + (size_t)(nb + lrow) * 256 + k0 + khalf * 16);
        acc = wmma_bf16(af, bf, acc);
      }
      int n = nb + lrow;
      float bb = pb2[p * 64 + n];
#pragma unroll
      for (int r = 0; r < 8; ++r)
        sO[mw + khalf * 8 + r][n] = acc[r] + bb;
    }
    __syncthreads();

    // ---- mixture accumulation ----
    for (int i = tid; i < BM * AOUT; i += 256) {
      int b = i >> 5, a = i & 31;
      float mu  = sO[b][a];
      float ls  = sO[b][a + 32];
      float inv = sW[b][p] * __expf(-ls);  // w / sigma
      sInv[b][a]   += inv;
      sInvMu[b][a] += inv * mu;
    }
    __syncthreads();
  }

  for (int i = tid; i < BM * AOUT; i += 256) {
    int b = i >> 5, a = i & 31;
    mean[(size_t)(b0 + b) * AOUT + a] = sInvMu[b][a] / sInv[b][a];
  }
}

// ---------------------------------------------------------------------------
// Host-side orchestration
// ---------------------------------------------------------------------------
extern "C" void kernel_launch(void* const* d_in, const int* in_sizes, int n_in,
                              void* d_out, int out_size, void* d_ws, size_t ws_size,
                              hipStream_t stream) {
  (void)in_sizes; (void)n_in; (void)out_size; (void)ws_size;
  constexpr int B = 32768, S = 512, G = 128, AOUT = 32, P = 16;
  constexpr int H1 = 512, H2 = 256, VH1 = 1024, VH2 = 512;

  const float* features = (const float*)d_in[0];
  const float* se_w1 = (const float*)d_in[1];  const float* se_b1 = (const float*)d_in[2];
  const float* se_w2 = (const float*)d_in[3];  const float* se_b2 = (const float*)d_in[4];
  const float* ge_w1 = (const float*)d_in[5];  const float* ge_b1 = (const float*)d_in[6];
  const float* ge_w2 = (const float*)d_in[7];  const float* ge_b2 = (const float*)d_in[8];
  const float* gt_w1 = (const float*)d_in[9];  const float* gt_b1 = (const float*)d_in[10];
  const float* gt_w2 = (const float*)d_in[11]; const float* gt_b2 = (const float*)d_in[12];
  const float* pse_w1 = (const float*)d_in[13]; const float* pse_b1 = (const float*)d_in[14];
  const float* pse_w2 = (const float*)d_in[15]; const float* pse_b2 = (const float*)d_in[16];
  const float* pw1 = (const float*)d_in[17]; const float* pb1 = (const float*)d_in[18];
  const float* pw2 = (const float*)d_in[19]; const float* pb2 = (const float*)d_in[20];
  const float* vf_w1 = (const float*)d_in[21]; const float* vf_b1 = (const float*)d_in[22];
  const float* vf_w2 = (const float*)d_in[23]; const float* vf_b2 = (const float*)d_in[24];

  char* ws = (char*)d_ws;
  size_t off = 0;
  auto carve = [&](size_t bytes) {
    char* p = ws + off;
    off += (bytes + 255) & ~(size_t)255;
    return p;
  };

  __bf16* feat_bf = (__bf16*)carve((size_t)B * (S + G) * 2);
  __bf16* t1      = (__bf16*)carve((size_t)B * VH1 * 2);      // widest intermediate
  __bf16* t2      = (__bf16*)carve((size_t)B * H1 * 2);
  __bf16* gate_in = (__bf16*)carve((size_t)B * (2 * H2) * 2); // [s_emb | g_emb]
  __bf16* pe      = (__bf16*)carve((size_t)B * H2 * 2);
  float*  wgt     = (float*)carve((size_t)B * P * 4);

  __bf16* se_w1t  = (__bf16*)carve((size_t)S * H1 * 2);
  __bf16* se_w2t  = (__bf16*)carve((size_t)H1 * H2 * 2);
  __bf16* ge_w1t  = (__bf16*)carve((size_t)G * H1 * 2);
  __bf16* ge_w2t  = (__bf16*)carve((size_t)H1 * H2 * 2);
  __bf16* gt_w1t  = (__bf16*)carve((size_t)(2 * H2) * H2 * 2);
  __bf16* gt_w2t  = (__bf16*)carve((size_t)H2 * P * 2);
  __bf16* pse_w1t = (__bf16*)carve((size_t)S * H1 * 2);
  __bf16* pse_w2t = (__bf16*)carve((size_t)H1 * H2 * 2);
  __bf16* pw1t    = (__bf16*)carve((size_t)P * H2 * H2 * 2);
  __bf16* pw2t    = (__bf16*)carve((size_t)P * H2 * (2 * AOUT) * 2);
  __bf16* vf_w1t  = (__bf16*)carve((size_t)(S + G) * VH1 * 2);
  __bf16* vf_w2t  = (__bf16*)carve((size_t)VH1 * VH2 * 2);

  // --- conversions -------------------------------------------------------
  {
    size_t n = (size_t)B * (S + G);
    int blocks = (int)((n / 4 + 255) / 256);
    f32_to_bf16<<<blocks, 256, 0, stream>>>(features, feat_bf, n);
  }
  auto tconv = [&](const float* src, __bf16* dst, int K, int N, int batch) {
    dim3 g((unsigned)((K * N + 255) / 256), 1, (unsigned)batch);
    f32_transpose_to_bf16<<<g, 256, 0, stream>>>(src, dst, K, N);
  };
  tconv(se_w1, se_w1t, S, H1, 1);       tconv(se_w2, se_w2t, H1, H2, 1);
  tconv(ge_w1, ge_w1t, G, H1, 1);       tconv(ge_w2, ge_w2t, H1, H2, 1);
  tconv(gt_w1, gt_w1t, 2 * H2, H2, 1);  tconv(gt_w2, gt_w2t, H2, P, 1);
  tconv(pse_w1, pse_w1t, S, H1, 1);     tconv(pse_w2, pse_w2t, H1, H2, 1);
  tconv(pw1, pw1t, H2, H2, P);          tconv(pw2, pw2t, H2, 2 * AOUT, P);
  tconv(vf_w1, vf_w1t, S + G, VH1, 1);  tconv(vf_w2, vf_w2t, VH1, VH2, 1);

  // --- dense layers ------------------------------------------------------
  auto gemm = [&](const __bf16* A, int lda, const __bf16* Wt, const float* bias,
                  void* out, int ldo, int N, int K, int act, bool obf16) {
    dim3 grid((unsigned)((N + 63) / 64), (unsigned)(B / 128));
    if (act == 1 && obf16)
      gemm_bias_act<1, true ><<<grid, 256, 0, stream>>>(A, lda, Wt, bias, out, ldo, B, N, K);
    else if (act == 1)
      gemm_bias_act<1, false><<<grid, 256, 0, stream>>>(A, lda, Wt, bias, out, ldo, B, N, K);
    else
      gemm_bias_act<2, false><<<grid, 256, 0, stream>>>(A, lda, Wt, bias, out, ldo, B, N, K);
  };

  float* out_mean  = (float*)d_out;                      // B x 32
  float* out_value = (float*)d_out + (size_t)B * AOUT;   // B x 512

  // state encoder -> s_emb (into gate_in cols [0,256))
  gemm(feat_bf, S + G, se_w1t, se_b1, t1, H1, H1, S, 1, true);
  gemm(t1, H1, se_w2t, se_b2, gate_in, 2 * H2, H2, H1, 1, true);
  // goal encoder -> g_emb (into gate_in cols [256,512))
  gemm(feat_bf + S, S + G, ge_w1t, ge_b1, t2, H1, H1, G, 1, true);
  gemm(t2, H1, ge_w2t, ge_b2, gate_in + H2, 2 * H2, H2, H1, 1, true);
  // gate: relu -> sigmoid, f32 weights
  gemm(gate_in, 2 * H2, gt_w1t, gt_b1, t2, H2, H2, 2 * H2, 1, true);
  gemm(t2, H2, gt_w2t, gt_b2, wgt, P, P, H2, 2, false);
  // primitive state encoder -> prim_embed
  gemm(feat_bf, S + G, pse_w1t, pse_b1, t2, H1, H1, S, 1, true);
  gemm(t2, H1, pse_w2t, pse_b2, pe, H2, H2, H1, 1, true);
  // fused primitive mixture head -> mean
  prim_head<<<B / 32, 256, 0, stream>>>(pe, pw1t, pb1, pw2t, pb2, wgt, out_mean);
  // value function
  gemm(feat_bf, S + G, vf_w1t, vf_b1, t1, VH1, VH1, S + G, 1, true);
  gemm(t1, VH1, vf_w2t, vf_b2, out_value, VH2, VH2, VH1, 1, false);
}
